// VertexMessagePass_77618648973581
// MI455X (gfx1250) — compile-verified, matrix operand order checked by reading, not run on hardware
//
#include <hip/hip_runtime.h>

typedef __attribute__((ext_vector_type(16))) _Float16 v16h;
typedef __attribute__((ext_vector_type(8)))  _Float16 v8h;
typedef __attribute__((ext_vector_type(8)))  float    v8f;

// Exact payload type for the async b128 builtin (per hipcc diagnostic: int __vector(4) *)
typedef int v4i_vs __attribute__((vector_size(16)));
typedef __attribute__((address_space(1))) v4i_vs* gb128_p;  // global (AS1) int4*
typedef __attribute__((address_space(3))) v4i_vs* lb128_p;  // LDS    (AS3) int4*

#define TILE_M   128          // vertices per workgroup
#define DD       256          // feature dim (fixed by reference)
#define WST      264          // weight LDS row stride in halves (pad 8 -> bank-conflict-free)
#define FST      264          // activation LDS row stride in halves
#define NTHREADS 256          // 8 wave32s

#if defined(__gfx1250__) && __has_builtin(__builtin_amdgcn_global_load_async_to_lds_b128)
#define HAVE_ASYNC_LDS 1
#else
#define HAVE_ASYNC_LDS 0
#endif

// Build a 16-half A/B fragment per ISA 7.12.2: elems 0..7 from [p], elems 8..15 from [p+16].
__device__ __forceinline__ v16h ld_frag(const _Float16* p) {
    union { v16h v; v8h h[2]; } u;
    u.h[0] = *(const v8h*)(p);
    u.h[1] = *(const v8h*)(p + 16);
    return u.v;
}

// Stage one 256x256 weight tile (f16, padded rows) into LDS.
// Fast path: flat copy of pre-packed f16 tile (async global->LDS when available).
// Fallback: convert from f32 source on the fly.
__device__ __forceinline__ void load_weight_tile(_Float16* sW, const _Float16* wpre, int stage,
                                                 const float* __restrict__ src,
                                                 int srcStride, int srcColOff, int tid)
{
    if (wpre) {
        const char* gsrc = (const char*)(wpre + (size_t)stage * (DD * WST));
#if HAVE_ASYNC_LDS
        for (int i = tid * 16; i < DD * WST * 2; i += NTHREADS * 16) {
            __builtin_amdgcn_global_load_async_to_lds_b128(
                (gb128_p)(gsrc + i),
                (lb128_p)((char*)sW + i),
                0, 0);
        }
#if __has_builtin(__builtin_amdgcn_s_wait_asynccnt)
        __builtin_amdgcn_s_wait_asynccnt(0);
#else
        asm volatile("s_wait_asynccnt 0" ::: "memory");
#endif
#else
        const uint4* s4 = (const uint4*)gsrc;
        uint4* d4 = (uint4*)sW;
        for (int i = tid; i < DD * WST * 2 / 16; i += NTHREADS) d4[i] = s4[i];
#endif
    } else {
        for (int i = tid; i < DD * DD; i += NTHREADS) {
            int e = i >> 8, d = i & 255;
            sW[e * WST + d] = (_Float16)src[(size_t)e * srcStride + srcColOff + d];
        }
    }
}

// One-shot prep: pack Wm, Wu[:, :256], Wu[:, 256:] as f16 tiles with padded (264-half) rows.
__global__ __launch_bounds__(256)
void pack_weights_kernel(const float* __restrict__ Wm, const float* __restrict__ Wu,
                         _Float16* __restrict__ wpre)
{
    int i = blockIdx.x * blockDim.x + threadIdx.x;
    const int per = DD * WST;
    if (i >= 3 * per) return;
    int stage = i / per;
    int r = i - stage * per;
    int e = r / WST, d = r - e * WST;
    float v = 0.0f;
    if (d < DD) {
        if (stage == 0)      v = Wm[e * DD + d];
        else if (stage == 1) v = Wu[(size_t)e * (2 * DD) + d];
        else                 v = Wu[(size_t)e * (2 * DD) + DD + d];
    }
    wpre[i] = (_Float16)v;
}

__global__ __launch_bounds__(NTHREADS)
void vmp_fused_kernel(const float* __restrict__ feats,
                      const int*   __restrict__ adj,
                      const float* __restrict__ Wm,
                      const float* __restrict__ bm,
                      const float* __restrict__ Wu,
                      const float* __restrict__ bu,
                      const _Float16* __restrict__ wpre,
                      float* __restrict__ out,
                      int N)
{
    extern __shared__ char smem_raw[];
    _Float16* sW   = (_Float16*)smem_raw;             // 256 x WST  (weight tile, f16)
    _Float16* sF   = sW + DD * WST;                   // 128 x FST  (own feats, f16)
    _Float16* sS   = sF + TILE_M * FST;               // 128 x FST  (neighbor sum -> agg, f16)
    int*      sAdj = (int*)(sS + TILE_M * FST);       // 128 x 3
    float*    sInv = (float*)(sAdj + 3 * TILE_M);     // 128  (1/count)

    const int tid = threadIdx.x;
    const int b   = blockIdx.y;
    const int v0  = blockIdx.x * TILE_M;
    const size_t batchBase = (size_t)b * (size_t)N * DD;

    // ---- stage 0: adjacency ----
    for (int i = tid; i < 3 * TILE_M; i += NTHREADS) {
        int v = v0 + i / 3;
        sAdj[i] = (v < N) ? adj[(size_t)v0 * 3 + i] : -1;
    }
    __syncthreads();

    if (tid < TILE_M) {
        int c = (sAdj[3*tid+0] >= 0) + (sAdj[3*tid+1] >= 0) + (sAdj[3*tid+2] >= 0);
        sInv[tid] = 1.0f / (float)(c < 1 ? 1 : c);
    }

    // ---- stage 1: gather own feats + fp32 neighbor sum, convert to f16 in LDS ----
    {
        const int c = tid;                    // NTHREADS == DD: one column per thread
        for (int r = 0; r < TILE_M; ++r) {
            int v = v0 + r;
            float f = 0.0f, s = 0.0f;
            if (v < N) {
                f = feats[batchBase + (size_t)v * DD + c];
                int i0 = sAdj[3*r+0], i1 = sAdj[3*r+1], i2 = sAdj[3*r+2];
                if (i0 >= 0) s += feats[batchBase + (size_t)i0 * DD + c];
                if (i1 >= 0) s += feats[batchBase + (size_t)i1 * DD + c];
                if (i2 >= 0) s += feats[batchBase + (size_t)i2 * DD + c];
            }
            sF[r * FST + c] = (_Float16)f;
            sS[r * FST + c] = (_Float16)s;
        }
    }

    // ---- stage 2: Wm tile -> LDS ----
    load_weight_tile(sW, wpre, 0, Wm, DD, 0, tid);
    __syncthreads();

    // wave-level WMMA addressing (ISA 7.12.2 wave32 layouts)
    const int w       = tid >> 5;             // wave id 0..7 -> 16-row stripe
    const int lane    = tid & 31;
    const int halfSel = lane >> 4;            // 0 | 1
    const int l15     = lane & 15;
    const int klo     = halfSel * 8;          // K sub-offset within 32-chunk
    const int rowA    = w * 16 + l15;         // A row (this lane's matrix row)
    const int rbC     = w * 16 + halfSel * 8; // C/D base row for this lane

    // ---- GEMM1: acc = S(16x256) x Wm^T -> agg ----
    v8f acc[16] = {};
    {
        const _Float16* aRow = sS + rowA * FST;
        #pragma unroll
        for (int kk = 0; kk < 8; ++kk) {
            v16h a = ld_frag(aRow + kk * 32 + klo);
            #pragma unroll
            for (int ct = 0; ct < 16; ++ct) {
                v16h bb = ld_frag(sW + (ct * 16 + l15) * WST + kk * 32 + klo);
                acc[ct] = __builtin_amdgcn_wmma_f32_16x16x32_f16(
                    false, a, false, bb, (short)0, acc[ct], false, false);
            }
        }
    }
    // scale by 1/count, add 3*bm/count, write agg (f16) over sS — rows are wave-local
    {
        float inv[8];
        #pragma unroll
        for (int i = 0; i < 8; ++i) inv[i] = sInv[rbC + i];
        #pragma unroll
        for (int ct = 0; ct < 16; ++ct) {
            int e = ct * 16 + l15;
            float bm3 = 3.0f * bm[e];
            #pragma unroll
            for (int i = 0; i < 8; ++i) {
                float val = (acc[ct][i] + bm3) * inv[i];
                sS[(rbC + i) * FST + e] = (_Float16)val;
            }
        }
    }
    __syncthreads();   // all waves done reading Wm

    // ---- GEMM2a: acc2 = F x Wu[:, :256]^T ----
    load_weight_tile(sW, wpre, 1, Wu, 2 * DD, 0, tid);
    __syncthreads();

    v8f acc2[16] = {};
    {
        const _Float16* aRow = sF + rowA * FST;
        #pragma unroll
        for (int kk = 0; kk < 8; ++kk) {
            v16h a = ld_frag(aRow + kk * 32 + klo);
            #pragma unroll
            for (int ct = 0; ct < 16; ++ct) {
                v16h bb = ld_frag(sW + (ct * 16 + l15) * WST + kk * 32 + klo);
                acc2[ct] = __builtin_amdgcn_wmma_f32_16x16x32_f16(
                    false, a, false, bb, (short)0, acc2[ct], false, false);
            }
        }
    }
    __syncthreads();   // all waves done reading Wu1

    // ---- GEMM2b: acc2 += agg x Wu[:, 256:]^T ----
    load_weight_tile(sW, wpre, 2, Wu, 2 * DD, DD, tid);
    __syncthreads();

    {
        const _Float16* aRow = sS + rowA * FST;
        #pragma unroll
        for (int kk = 0; kk < 8; ++kk) {
            v16h a = ld_frag(aRow + kk * 32 + klo);
            #pragma unroll
            for (int ct = 0; ct < 16; ++ct) {
                v16h bb = ld_frag(sW + (ct * 16 + l15) * WST + kk * 32 + klo);
                acc2[ct] = __builtin_amdgcn_wmma_f32_16x16x32_f16(
                    false, a, false, bb, (short)0, acc2[ct], false, false);
            }
        }
    }

    // ---- epilogue: + bu, store fp32 output ----
    #pragma unroll
    for (int ct = 0; ct < 16; ++ct) {
        int e = ct * 16 + l15;
        float buv = bu[e];
        #pragma unroll
        for (int i = 0; i < 8; ++i) {
            int r = rbC + i;
            int v = v0 + r;
            if (v < N) out[batchBase + (size_t)v * DD + e] = acc2[ct][i] + buv;
        }
    }
}

extern "C" void kernel_launch(void* const* d_in, const int* in_sizes, int n_in,
                              void* d_out, int out_size, void* d_ws, size_t ws_size,
                              hipStream_t stream) {
    const float* feats = (const float*)d_in[0];
    const int*   adj   = (const int*)  d_in[1];
    const float* Wm    = (const float*)d_in[2];
    const float* bm    = (const float*)d_in[3];
    const float* Wu    = (const float*)d_in[4];
    const float* bu    = (const float*)d_in[5];
    float* out = (float*)d_out;

    const int N = in_sizes[1] / 3;                  // vertex_adj is (N, 3)
    const int B = (int)(in_sizes[0] / ((size_t)N * DD));
    const int tiles = (N + TILE_M - 1) / TILE_M;

    // Pre-packed f16 weight tiles in workspace (3 x 256 x 264 halves ~ 396 KB)
    const size_t wpre_bytes = (size_t)3 * DD * WST * 2;
    _Float16* wpre = nullptr;
    if (ws_size >= wpre_bytes) {
        wpre = (_Float16*)d_ws;
        const int total = 3 * DD * WST;
        pack_weights_kernel<<<(total + 255) / 256, 256, 0, stream>>>(Wm, Wu, wpre);
    }

    const size_t smem = (size_t)(DD * WST + 2 * TILE_M * FST) * 2  // f16 buffers
                      + (size_t)(3 * TILE_M) * sizeof(int)
                      + (size_t)TILE_M * sizeof(float);            // ~266 KB of 320 KB LDS

    dim3 grid(tiles, B);
    vmp_fused_kernel<<<grid, NTHREADS, smem, stream>>>(feats, adj, Wm, bm, Wu, bu, wpre, out, N);
}
